// NeuralODEModel_66236985639756
// MI455X (gfx1250) — matrix operand order repaired and checked by reading, not compiled, not run on hardware
//
#include <hip/hip_runtime.h>
#include <math.h>

typedef __attribute__((ext_vector_type(16))) _Float16 v16h;
typedef __attribute__((ext_vector_type(8)))  float    v8f;

union AFrag {
    v16h     v;
    _Float16 h[16];
    unsigned u[8];
};

__device__ __forceinline__ void lds_fence() {
    // LDS ops from one wave complete in order; wait + compiler memory barrier
    asm volatile("s_wait_dscnt 0" ::: "memory");
}

__device__ __forceinline__ float fast_tanh(float x) {
#if __has_builtin(__builtin_amdgcn_tanhf)
    return __builtin_amdgcn_tanhf(x);      // v_tanh_f32 (gfx1250 TRANS op)
#else
    float e = __expf(2.0f * x);            // v_exp_f32 path
    return 1.0f - 2.0f / (e + 1.0f);
#endif
}

#define WMMA_F16(A, B, C) \
    __builtin_amdgcn_wmma_f32_16x16x32_f16(false, (A), false, (B), (short)0, (C), false, false)

// per-wave LDS layout (f16 staging buffers, padded strides to dodge bank conflicts)
constexpr int HS = 18;                       // h buffer row stride (halves)
constexpr int AS = 66;                       // a1 buffer row stride (halves)
constexpr int HBUF_BYTES = 16 * HS * 2;      // 576
constexpr int ABUF_BYTES = 16 * AS * 2;      // 2112
constexpr int WAVE_LDS   = 2816;             // 576 + 2112 + 64(dose) + pad
constexpr int WAVES      = 8;                // waves per block (256 threads)

__global__ __launch_bounds__(256, 1) void node_dopri5_kernel(
    const float* __restrict__ tarr,
    const float* __restrict__ sfeat,   // [B,7]
    const float* __restrict__ itime,   // [B,8]
    const float* __restrict__ idose,   // [B,8]
    const float* __restrict__ Wse, const float* __restrict__ bse,   // 7x16, 16
    const float* __restrict__ Wi,  const float* __restrict__ bi,    // 16x16, 16
    const float* __restrict__ W1,  const float* __restrict__ b1,    // 33x64, 64
    const float* __restrict__ W2,  const float* __restrict__ b2,    // 64x16, 16
    const float* __restrict__ Wr,  const float* __restrict__ br,    // 16x1, 1
    float* __restrict__ out, int T, int Btot)
{
    __shared__ __align__(16) unsigned char smem[WAVES * WAVE_LDS];

    const int lane = threadIdx.x & 31;
    const int wid  = threadIdx.x >> 5;
    const int nloc = lane & 15;              // column / N index inside tile
    const int hi   = lane >> 4;              // half-wave select
    const int tile = blockIdx.x * WAVES + wid;
    const int b0   = tile * 16;
    if (b0 + 16 > Btot) return;

    _Float16* hbuf    = (_Float16*)(smem + wid * WAVE_LDS);
    _Float16* a1buf   = (_Float16*)(smem + wid * WAVE_LDS + HBUF_BYTES);
    float*    dosebuf = (float*)   (smem + wid * WAVE_LDS + HBUF_BYTES + ABUF_BYTES);
    float*    sebuf   = (float*)a1buf;       // init-time overlay (16x16 f32 = 1024B)

    // ---------------- load loop-invariant weights into B-fragments --------------
    // B-frag layout (16-bit, KxN=32x16): lane l holds column n=l%16,
    // half i = W[8*(l/16) + (i&7) + 16*(i>>3)][n] (matches A layout).
    const int kb16 = 16 * hi;
    AFrag B1[4], B2[2];
    #pragma unroll
    for (int n = 0; n < 4; ++n)
        #pragma unroll
        for (int i = 0; i < 16; ++i)
            B1[n].h[i] = (_Float16)W1[(8 * hi + (i & 7) + 16 * (i >> 3)) * 64 + 16 * n + nloc];
    #pragma unroll
    for (int t2 = 0; t2 < 2; ++t2)
        #pragma unroll
        for (int i = 0; i < 16; ++i)
            B2[t2].h[i] = (_Float16)W2[(32 * t2 + 8 * hi + (i & 7) + 16 * (i >> 3)) * 16 + nloc];

    float b1n[4], w1cn[4];
    #pragma unroll
    for (int n = 0; n < 4; ++n) {
        b1n[n]  = b1[16 * n + nloc];
        w1cn[n] = W1[32 * 64 + 16 * n + nloc];  // dose row of W1
    }
    const float b2v     = b2[nloc];
    const float wr_lane = Wr[nloc];
    const float br0     = br[0];

    float tj[8], dj[8];
    #pragma unroll
    for (int n = 0; n < 8; ++n) {
        tj[n] = itime[(b0 + nloc) * 8 + n];
        dj[n] = idose[(b0 + nloc) * 8 + n];
    }

    // ---------------- static encoder + h0 (D-layout: vgpr j -> row j+8*hi) ------
    float seD[8];
    #pragma unroll
    for (int j = 0; j < 8; ++j) {
        const int m = j + 8 * hi;
        const float* row = sfeat + (size_t)(b0 + m) * 7;
        float acc = bse[nloc];
        #pragma unroll
        for (int i = 0; i < 7; ++i) acc = fmaf(row[i], Wse[i * 16 + nloc], acc);
        seD[j] = fmaxf(acc, 0.0f);
    }
    #pragma unroll
    for (int j = 0; j < 8; ++j) sebuf[(j + 8 * hi) * 16 + nloc] = seD[j];
    lds_fence();

    v8f h;
    #pragma unroll
    for (int j = 0; j < 8; ++j) {
        const int m = j + 8 * hi;
        float acc = bi[nloc];
        #pragma unroll
        for (int k = 0; k < 16; ++k) acc = fmaf(sebuf[m * 16 + k], Wi[k * 16 + nloc], acc);
        h[j] = acc;
    }

    // A-frag static half: inp features 16..31 = static_embed; loop-invariant.
    // A layout (16x32): lane l holds row m=l%16, halves 0..7 = K kb..kb+7,
    // halves 8..15 = K 16+kb..16+kb+7, with kb = 8*(l/16).
    AFrag aH;
    {
        const int kb = 8 * hi;
        #pragma unroll
        for (int i = 0; i < 8; ++i)
            aH.h[8 + i] = (_Float16)sebuf[nloc * 16 + kb + i];
    }
    lds_fence();   // sebuf overlays a1buf; done reading before odef writes it

    // ---------------- ODE right-hand side -------------------------------------
    auto odef = [&](v8f y, float tv) -> v8f {
        // stage state -> LDS (f16), D-layout scatter
        #pragma unroll
        for (int j = 0; j < 8; ++j)
            hbuf[(j + 8 * hi) * HS + nloc] = (_Float16)y[j];

        // per-batch dose (lane's batch = b0 + nloc)
        float d = 0.0f;
        #pragma unroll
        for (int n = 0; n < 8; ++n) {
            float u = tv - tj[n];
            d = fmaf(dj[n], __expf(-0.5f * u * u), d);
        }
        d = fminf(fmaxf(d, 0.0f), 100.0f);
        if (hi == 0) dosebuf[nloc] = d;
        lds_fence();

        // refresh h half of layer-1 A fragment (gather transposed rows)
        {
            const unsigned* hrow = (const unsigned*)(hbuf + nloc * HS + 8 * hi);
            #pragma unroll
            for (int v = 0; v < 4; ++v) aH.u[v] = hrow[v];
        }
        float dl[8];
        #pragma unroll
        for (int j = 0; j < 8; ++j) dl[j] = dosebuf[j + 8 * hi];

        // layer 1: z1 = [h|s] @ W1[0:32] + (b1 + dose*W1[32]); tanh; stage to LDS
        #pragma unroll
        for (int n = 0; n < 4; ++n) {
            v8f c;
            #pragma unroll
            for (int j = 0; j < 8; ++j) c[j] = fmaf(dl[j], w1cn[n], b1n[n]);
            c = WMMA_F16(aH.v, B1[n].v, c);
            #pragma unroll
            for (int j = 0; j < 8; ++j) {
                float z = fast_tanh(c[j]);
                a1buf[(j + 8 * hi) * AS + 16 * n + nloc] = (_Float16)z;
            }
        }
        lds_fence();

        // layer 2: k = tanh(z1) @ W2 + b2   (two K=32 chained WMMAs)
        v8f k;
        #pragma unroll
        for (int j = 0; j < 8; ++j) k[j] = b2v;
        const _Float16* arow = a1buf + nloc * AS;
        const int kb = 8 * hi;
        #pragma unroll
        for (int t2 = 0; t2 < 2; ++t2) {
            AFrag A2;
            #pragma unroll
            for (int v = 0; v < 4; ++v) {
                A2.u[v]     = *(const unsigned*)(arow + 32 * t2 + kb + 2 * v);
                A2.u[4 + v] = *(const unsigned*)(arow + 32 * t2 + 16 + kb + 2 * v);
            }
            k = WMMA_F16(A2.v, B2[t2].v, k);
        }
        return k;
    };

    // ---------------- one fixed dopri5 step -----------------------------------
    auto rk_step = [&](v8f hc, float tv, float dt) -> v8f {
        v8f k1 = odef(hc, tv);
        v8f y;
        #pragma unroll
        for (int j = 0; j < 8; ++j) y[j] = fmaf(dt, 0.2f * k1[j], hc[j]);
        v8f k2 = odef(y, tv + 0.2f * dt);
        #pragma unroll
        for (int j = 0; j < 8; ++j)
            y[j] = fmaf(dt, fmaf(3.0f/40.0f, k1[j], (9.0f/40.0f) * k2[j]), hc[j]);
        v8f k3 = odef(y, tv + 0.3f * dt);
        #pragma unroll
        for (int j = 0; j < 8; ++j) {
            float s = fmaf(44.0f/45.0f, k1[j],
                      fmaf(-56.0f/15.0f, k2[j], (32.0f/9.0f) * k3[j]));
            y[j] = fmaf(dt, s, hc[j]);
        }
        v8f k4 = odef(y, tv + 0.8f * dt);
        #pragma unroll
        for (int j = 0; j < 8; ++j) {
            float s = fmaf(19372.0f/6561.0f, k1[j],
                      fmaf(-25360.0f/2187.0f, k2[j],
                      fmaf(64448.0f/6561.0f, k3[j], (-212.0f/729.0f) * k4[j])));
            y[j] = fmaf(dt, s, hc[j]);
        }
        v8f k5 = odef(y, tv + (8.0f/9.0f) * dt);
        #pragma unroll
        for (int j = 0; j < 8; ++j) {
            float s = fmaf(9017.0f/3168.0f, k1[j],
                      fmaf(-355.0f/33.0f, k2[j],
                      fmaf(46732.0f/5247.0f, k3[j],
                      fmaf(49.0f/176.0f, k4[j], (-5103.0f/18656.0f) * k5[j]))));
            y[j] = fmaf(dt, s, hc[j]);
        }
        v8f k6 = odef(y, tv + dt);
        v8f hn;
        #pragma unroll
        for (int j = 0; j < 8; ++j) {
            float s = fmaf(35.0f/384.0f, k1[j],
                      fmaf(500.0f/1113.0f, k3[j],
                      fmaf(125.0f/192.0f, k4[j],
                      fmaf(-2187.0f/6784.0f, k5[j], (11.0f/84.0f) * k6[j]))));
            hn[j] = fmaf(dt, s, hc[j]);
        }
        return hn;
    };

    // ---------------- readout: relu(h @ Wr + br), 16-lane butterfly reduce ----
    auto readout = [&](v8f hc, int t) {
        #pragma unroll
        for (int j = 0; j < 8; ++j) {
            float p = hc[j] * wr_lane;
            p += __shfl_xor(p, 1);
            p += __shfl_xor(p, 2);
            p += __shfl_xor(p, 4);
            p += __shfl_xor(p, 8);
            if (nloc == 0)
                out[(size_t)(b0 + j + 8 * hi) * T + t] = fmaxf(p + br0, 0.0f);
        }
    };

    readout(h, 0);

    #pragma unroll 1
    for (int iv = 0; iv < T - 1; ++iv) {
        const float t0 = tarr[iv];
        const float t1 = tarr[iv + 1];
        const float dt = (t1 - t0) * 0.5f;   // SUBSTEPS = 2
        #pragma unroll 1
        for (int ss = 0; ss < 2; ++ss)
            h = rk_step(h, fmaf((float)ss, dt, t0), dt);
        readout(h, iv + 1);
    }
}

extern "C" void kernel_launch(void* const* d_in, const int* in_sizes, int n_in,
                              void* d_out, int out_size, void* d_ws, size_t ws_size,
                              hipStream_t stream) {
    (void)n_in; (void)out_size; (void)d_ws; (void)ws_size;
    const float* tarr = (const float*)d_in[0];
    const float* sf   = (const float*)d_in[1];
    const float* it   = (const float*)d_in[2];
    const float* idz  = (const float*)d_in[3];
    const float* Wse  = (const float*)d_in[4];
    const float* bse  = (const float*)d_in[5];
    const float* Wi   = (const float*)d_in[6];
    const float* bi   = (const float*)d_in[7];
    const float* W1   = (const float*)d_in[8];
    const float* b1   = (const float*)d_in[9];
    const float* W2   = (const float*)d_in[10];
    const float* b2   = (const float*)d_in[11];
    const float* Wr   = (const float*)d_in[12];
    const float* br   = (const float*)d_in[13];

    const int T = in_sizes[0];           // 64
    const int B = in_sizes[1] / 7;       // 32768
    const int tiles  = (B + 15) / 16;    // 2048 waves
    const int blocks = (tiles + WAVES - 1) / WAVES;

    node_dopri5_kernel<<<blocks, 32 * WAVES, 0, stream>>>(
        tarr, sf, it, idz, Wse, bse, Wi, bi, W1, b1, W2, b2, Wr, br,
        (float*)d_out, T, B);
}
